// Kernel_13538758526885243862_65652870087083
// MI455X (gfx1250) — matrix operand order repaired and checked by reading, not compiled, not run on hardware
//
#include <hip/hip_runtime.h>

#define USE_ASYNC_B 1

typedef __attribute__((ext_vector_type(16))) _Float16 v16h;
typedef __attribute__((ext_vector_type(8)))  _Float16 v8h;
typedef __attribute__((ext_vector_type(2)))  _Float16 h2;
typedef __attribute__((ext_vector_type(8)))  float    v8f;

constexpr int NB = 32;       // batch
constexpr int C  = 256;
constexpr int C8 = 32;
constexpr int H  = 56;
constexpr int W  = 56;
constexpr int HW = H * W;
constexpr int M  = NB * HW;  // 100352 GEMM rows (spatial positions)
static_assert(HW % 16 == 0, "16-row M-tile must not straddle a batch image");

// Clamped, branch-free im2col fetch (always-safe address + select-to-zero).
template<typename TIN>
__device__ __forceinline__ _Float16 ld_clamp(const TIN* __restrict__ base,
                                             size_t co, int xx, int yy)
{
    const bool inb = ((unsigned)xx < (unsigned)W) & ((unsigned)yy < (unsigned)H);
    const TIN v = base[co + (size_t)(inb ? yy : 0) * W + (inb ? xx : 0)];
    return inb ? (_Float16)v : (_Float16)0;
}

// ---------------------------------------------------------------------------
// Dilated-conv-as-GEMM on WMMA, double-buffered LDS staging, f16 weights.
// out[n,o,y,x] = sum_{c<CIN} sum_{t<T} in[n,c, y+(t-T/2)*DILY, x+(t-T/2)*DILX]
//                                      * wgt[o, c*T + t]          (zero pad)
// Block = 8 waves laid out WM x WO (WM*WO==8); each wave computes one
// 16(M) x 32(O) tile (two accumulators sharing one A fragment).
// Per 32-wide K chunk: A panel gathered+converted to LDS (packed b32 stores);
// B panel async-copied global->LDS (weights pre-converted to f16).
// ---------------------------------------------------------------------------
template<int CIN, int T, int O, int DILX, int DILY, int WM,
         typename TIN, typename TOUT>
__global__ __launch_bounds__(256)
void conv_gemm_wmma(const TIN* __restrict__ in, const _Float16* __restrict__ wgt,
                    TOUT* __restrict__ out)
{
    constexpr int K    = CIN * T;          // 768 / 96 / 224 (all % 32 == 0)
    constexpr int OTT  = O / 16;           // total o-tiles
    constexpr int WO   = 8 / WM;           // wave columns
    constexpr int OC   = WO * 2;           // o-tiles per block
    constexpr int AR   = WM * 16;          // A panel rows
    constexpr int BR   = OC * 16;          // B panel rows (output channels)
    constexpr int OBLK = OTT / OC;         // blocks along O
    constexpr int AJ   = AR / 16;          // A pair-stores per thread
    constexpr int BJ   = BR / 16;          // B b32 words per thread
    static_assert(OTT % OC == 0 && M % AR == 0, "tiling must divide");

    __shared__ _Float16 As[2][AR * 32];
    __shared__ _Float16 Bs[2][BR * 32];

    const int tid  = threadIdx.x;
    const int lane = tid & 31;
    const int wv   = tid >> 5;
    const int wmv  = wv / WO;
    const int wov  = wv - wmv * WO;
    const int hi   = lane >> 4;
    const int l15  = lane & 15;

    const int bo = blockIdx.x % OBLK;
    const int bm = blockIdx.x / OBLK;
    const int m_base = bm * AR;
    const int o_base = bo * BR;

    // Staging decomposition: element pair p = tid + 256*j covers
    // (row = tid>>4 + 16j, k = 2*(tid&15) + {0,1}).
    const int kp  = tid & 15;
    const int ra0 = tid >> 4;
    const TIN* abase[AJ];
    int axr[AJ], ayr[AJ];
#pragma unroll
    for (int j = 0; j < AJ; ++j) {
        const int m = m_base + ra0 + 16 * j;
        const int n = m / HW;
        const int r = m - n * HW;
        ayr[j] = r / W;
        axr[j] = r - ayr[j] * W;
        abase[j] = in + (size_t)n * CIN * HW;
    }

    v8f acc0 = {}, acc1 = {};
    int buf = 0;
    for (int kk0 = 0; kk0 < K; kk0 += 32, buf ^= 1) {
        // ---- stage A panel chunk (two taps per thread, packed b32 store) ----
        const int k0  = kk0 + 2 * kp;
        const int c0  = k0 / T,      t0 = k0 - c0 * T;
        const int c1  = (k0 + 1) / T, t1v = (k0 + 1) - c1 * T;
        const int dx0 = (t0 - T / 2) * DILX,  dy0 = (t0 - T / 2) * DILY;
        const int dx1 = (t1v - T / 2) * DILX, dy1 = (t1v - T / 2) * DILY;
        const size_t co0 = (size_t)c0 * HW, co1 = (size_t)c1 * HW;
#pragma unroll
        for (int j = 0; j < AJ; ++j) {
            h2 p;
            p[0] = ld_clamp(abase[j], co0, axr[j] + dx0, ayr[j] + dy0);
            p[1] = ld_clamp(abase[j], co1, axr[j] + dx1, ayr[j] + dy1);
            *(h2*)&As[buf][(ra0 + 16 * j) * 32 + 2 * kp] = p;
        }
        // ---- stage B panel chunk (f16 weights; 64B contiguous per row) ----
#if USE_ASYNC_B
        const unsigned bs_base = (unsigned)(uintptr_t)&Bs[buf][0];
        const unsigned wbase   = ((unsigned)o_base * K + (unsigned)kk0) * 2u;
#pragma unroll
        for (int j = 0; j < BJ; ++j) {
            const int rb = ra0 + 16 * j;
            const unsigned dst  = bs_base + (unsigned)(rb * 64 + kp * 4);
            const unsigned goff = wbase + (unsigned)rb * (K * 2) + kp * 4;
            asm volatile("global_load_async_to_lds_b32 %0, %1, %2"
                         :: "v"(dst), "v"(goff), "s"(wgt) : "memory");
        }
        asm volatile("s_wait_asynccnt 0" ::: "memory");
#else
        const _Float16* wp = wgt + (size_t)o_base * K + kk0 + 2 * kp;
#pragma unroll
        for (int j = 0; j < BJ; ++j) {
            const int rb = ra0 + 16 * j;
            *(h2*)&Bs[buf][rb * 32 + 2 * kp] = *(const h2*)&wp[(size_t)rb * K];
        }
#endif
        __syncthreads();

        // ---- fragments from LDS ----
        // A fragment, this kernel's contiguous k order: lane half hi -> k 16hi..16hi+15
        const v8h* pa = (const v8h*)&As[buf][(wmv * 16 + l15) * 32 + hi * 16];
        const v16h a = __builtin_shufflevector(pa[0], pa[1],
            0,1,2,3,4,5,6,7,8,9,10,11,12,13,14,15);
        // B fragment, same contiguous k order as A: lane half hi -> k rows 16hi..16hi+15
        const v8h* pb0 = (const v8h*)&Bs[buf][((wov * 2 + 0) * 16 + l15) * 32 + hi * 16];
        const v16h b0 = __builtin_shufflevector(pb0[0], pb0[1],
            0,1,2,3,4,5,6,7,8,9,10,11,12,13,14,15);
        const v8h* pb1 = (const v8h*)&Bs[buf][((wov * 2 + 1) * 16 + l15) * 32 + hi * 16];
        const v16h b1 = __builtin_shufflevector(pb1[0], pb1[1],
            0,1,2,3,4,5,6,7,8,9,10,11,12,13,14,15);

        acc0 = __builtin_amdgcn_wmma_f32_16x16x32_f16(false, a, false, b0,
                                                      (short)0, acc0, false, false);
        acc1 = __builtin_amdgcn_wmma_f32_16x16x32_f16(false, a, false, b1,
                                                      (short)0, acc1, false, false);
    }

    // C layout: lane -> N = oc ; VGPR r -> M = m0 + r + 8*hi.
    // HW % 16 == 0, so the whole 16-row tile lives in one batch image:
    // one division instead of eight.
    const int m0  = m_base + wmv * 16;
    const int nn  = m0 / HW;
    const int rm0 = m0 - nn * HW + (hi << 3);
    const int ocA = o_base + (wov * 2 + 0) * 16 + l15;
    const int ocB = ocA + 16;
    TOUT* outA = out + (size_t)nn * O * HW + (size_t)ocA * HW + rm0;
    TOUT* outB = out + (size_t)nn * O * HW + (size_t)ocB * HW + rm0;
#pragma unroll
    for (int r = 0; r < 8; ++r) {
        outA[r] = (TOUT)acc0[r];
        outB[r] = (TOUT)acc1[r];
    }
}

// ---------------------------------------------------------------------------
__global__ __launch_bounds__(256)
void cvt_f32_to_f16(const float* __restrict__ s, _Float16* __restrict__ d, int n)
{
    const int i = blockIdx.x * 256 + threadIdx.x;
    if (i < n) d[i] = (_Float16)s[i];
}

// ---------------------------------------------------------------------------
// t5 = p2w[ch] * depthwise_conv_H(t1, w5, dil 3, pad 3)   (f16 output)
// ---------------------------------------------------------------------------
__global__ __launch_bounds__(256)
void t5_kernel(const float* __restrict__ t1, const float* __restrict__ p2w,
               const float* __restrict__ w5, _Float16* __restrict__ t5)
{
    const int idx = blockIdx.x * 256 + threadIdx.x;      // N*C8*HW total
    if (idx >= NB * C8 * HW) return;
    const int xw = idx % W;
    int tmp = idx / W;
    const int y  = tmp % H;  tmp /= H;
    const int ch = tmp % C8;
    const int n  = tmp / C8;
    const float* base = t1 + ((size_t)(n * C8 + ch)) * HW;
    float s = 0.0f;
#pragma unroll
    for (int k = 0; k < 3; ++k) {
        const int yy = y + 3 * k - 3;
        const bool inb = (unsigned)yy < (unsigned)H;
        const float v = base[(inb ? yy : 0) * W + xw];
        s += inb ? w5[ch * 3 + k] * v : 0.0f;
    }
    t5[idx] = (_Float16)(p2w[ch] * s);
}

// ---------------------------------------------------------------------------
// t4[n,y,x] = sum_c sum_t t3[n,c,y,x+3t-3] * w4[c,t]   (256 -> 1 reduction)
// ---------------------------------------------------------------------------
__global__ __launch_bounds__(256)
void t4_kernel(const float* __restrict__ t3, const float* __restrict__ w4,
               float* __restrict__ t4o)
{
    const int idx = blockIdx.x * 256 + threadIdx.x;      // N*HW total
    if (idx >= NB * HW) return;
    const int xw = idx % W;
    int tmp = idx / W;
    const int y = tmp % H;
    const int n = tmp / H;
    float acc = 0.0f;
    for (int c = 0; c < C; ++c) {
        const float* row = t3 + ((size_t)(n * C + c)) * HW + y * W;
#pragma unroll
        for (int t = 0; t < 3; ++t) {
            const int xx = xw + 3 * t - 3;
            const bool inb = (unsigned)xx < (unsigned)W;
            const float v = row[inb ? xx : 0];
            acc += inb ? v * w4[c * 3 + t] : 0.0f;
        }
    }
    t4o[idx] = acc;
}

// ---------------------------------------------------------------------------
// t11: 7x1 (vertical) dilated(3) max over zero-padded t4, relu, 3-row sum of the 7
//      column-shift channels, /21.
// ---------------------------------------------------------------------------
__global__ __launch_bounds__(256)
void pool_kernel(const float* __restrict__ t4, float* __restrict__ t11)
{
    const int idx = blockIdx.x * 256 + threadIdx.x;      // N*HW total
    if (idx >= NB * HW) return;
    const int xw = idx % W;
    int tmp = idx / W;
    const int y = tmp % H;
    const int n = tmp / H;
    const float* b = t4 + (size_t)n * HW;
    float acc = 0.0f;
    for (int j = 0; j < 7; ++j) {
        const int xx = xw + 3 * j - 9;
        const bool inx = (unsigned)xx < (unsigned)W;
#pragma unroll
        for (int d = -1; d <= 1; ++d) {
            const int yc = y + d;
            if ((unsigned)yc >= (unsigned)H) continue;   // pad-after-relu -> 0
            float mx = -3.4e38f;
#pragma unroll
            for (int i = 0; i < 7; ++i) {
                const int yy = yc + 3 * i - 9;
                const bool inb = inx & ((unsigned)yy < (unsigned)H);
                float v = b[(inb ? yy : 0) * W + (inx ? xx : 0)];
                v = inb ? v : 0.0f;                      // zero pad BEFORE max
                mx = fmaxf(mx, v);
            }
            acc += fmaxf(mx, 0.0f);                      // relu
        }
    }
    t11[idx] = acc * (1.0f / 21.0f);
}

// ---------------------------------------------------------------------------
// t14: 3x3 conv, single channel, zero pad 1
// ---------------------------------------------------------------------------
__global__ __launch_bounds__(256)
void t14_kernel(const float* __restrict__ t11, const float* __restrict__ w14,
                float* __restrict__ t14o)
{
    const int idx = blockIdx.x * 256 + threadIdx.x;      // N*HW total
    if (idx >= NB * HW) return;
    const int xw = idx % W;
    int tmp = idx / W;
    const int y = tmp % H;
    const int n = tmp / H;
    const float* b = t11 + (size_t)n * HW;
    float acc = 0.0f;
#pragma unroll
    for (int a = 0; a < 3; ++a) {
#pragma unroll
        for (int c2 = 0; c2 < 3; ++c2) {
            const int yy = y + a - 1, xx = xw + c2 - 1;
            const bool inb = ((unsigned)yy < (unsigned)H) &
                             ((unsigned)xx < (unsigned)W);
            const float v = b[(inb ? yy : 0) * W + (inb ? xx : 0)];
            acc += inb ? w14[a * 3 + c2] * v : 0.0f;
        }
    }
    t14o[idx] = acc;
}

// ---------------------------------------------------------------------------
// out = t14 (broadcast over channels) + depthwise_conv_W(t15, w16, dil 3)
// ---------------------------------------------------------------------------
__global__ __launch_bounds__(256)
void final_kernel(const float* __restrict__ t15, const float* __restrict__ w16,
                  const float* __restrict__ t14v, float* __restrict__ out)
{
    const int idx = blockIdx.x * 256 + threadIdx.x;      // N*C*HW total
    if (idx >= NB * C * HW) return;
    const int xw = idx % W;
    int tmp = idx / W;
    const int y = tmp % H;
    const int ch = tmp / H;                              // n*C + o
    const int o  = ch % C;
    const int n  = ch / C;
    const float* row = t15 + (size_t)ch * HW + y * W;
    float acc = t14v[(size_t)n * HW + y * W + xw];
#pragma unroll
    for (int k = 0; k < 3; ++k) {
        const int xx = xw + 3 * k - 3;
        const bool inb = (unsigned)xx < (unsigned)W;
        const float v = row[inb ? xx : 0];
        acc += inb ? w16[o * 3 + k] * v : 0.0f;
    }
    out[idx] = acc;
}

// ---------------------------------------------------------------------------
extern "C" void kernel_launch(void* const* d_in, const int* in_sizes, int n_in,
                              void* d_out, int out_size, void* d_ws, size_t ws_size,
                              hipStream_t stream)
{
    const float* x    = (const float*)d_in[0];
    const float* w1   = (const float*)d_in[1];
    const float* p2w  = (const float*)d_in[2];
    const float* w3   = (const float*)d_in[3];
    const float* w4   = (const float*)d_in[4];
    const float* w5   = (const float*)d_in[5];
    const float* w8   = (const float*)d_in[6];
    const float* w12  = (const float*)d_in[7];
    const float* w14  = (const float*)d_in[8];
    const float* w15  = (const float*)d_in[9];
    const float* w16  = (const float*)d_in[10];
    float* out = (float*)d_out;

    // Workspace carve-up (float slots). t8h aliases t1, t12h aliases t3.
    float* ws = (float*)d_ws;
    size_t off = 0;
    float*     t1   = ws + off; off += (size_t)NB * C8 * HW;       // f32
    float*     t3   = ws + off; off += (size_t)NB * C  * HW;       // f32
    float*     t4   = ws + off; off += (size_t)NB * HW;
    _Float16*  t5h  = (_Float16*)(ws + off); off += (size_t)NB * C8 * HW / 2;
    float*     t11  = ws + off; off += (size_t)NB * HW;
    float*     t14  = ws + off; off += (size_t)NB * HW;
    float*     t15  = ws + off; off += (size_t)NB * C  * HW;       // f32
    _Float16*  wh   = (_Float16*)(ws + off);                        // f16 weights
    _Float16*  w1h  = wh;                 // 32*768   = 24576
    _Float16*  w3h  = w1h + 32 * 768;     // 256*96   = 24576
    _Float16*  w8h  = w3h + 256 * 96;     // 32*224   = 7168
    _Float16*  w12h = w8h + 32 * 224;     // 256*224  = 57344
    _Float16*  w15h = w12h + 256 * 224;   // 256*768  = 196608
    _Float16*  t8h  = (_Float16*)t1;      // t1 dead after t3 & t5
    _Float16*  t12h = (_Float16*)t3;      // t3 dead after t4

    // Pre-convert weights to f16 (tiny).
    cvt_f32_to_f16<<<(32 * 768 + 255) / 256, 256, 0, stream>>>(w1, w1h, 32 * 768);
    cvt_f32_to_f16<<<(256 * 96 + 255) / 256, 256, 0, stream>>>(w3, w3h, 256 * 96);
    cvt_f32_to_f16<<<(32 * 224 + 255) / 256, 256, 0, stream>>>(w8, w8h, 32 * 224);
    cvt_f32_to_f16<<<(256 * 224 + 255) / 256, 256, 0, stream>>>(w12, w12h, 256 * 224);
    cvt_f32_to_f16<<<(256 * 768 + 255) / 256, 256, 0, stream>>>(w15, w15h, 256 * 768);

    // O=32 stages: WM=8 (block: 128 rows x all 32 outputs), OBLK=1 -> 784 blocks
    // O=256 stages: WM=2 (block: 32 rows x 128 outputs),   OBLK=2 -> 6272 blocks
    constexpr int GRID_O32  = M / 128;
    constexpr int GRID_O256 = (M / 32) * 2;

    // 1) t1 = conv(x, w1): 256ch, 3 width taps dil 3 -> 32ch (f32 out: 2 users)
    conv_gemm_wmma<C, 3, C8, 3, 0, 8, float, float>
        <<<GRID_O32, 256, 0, stream>>>(x, w1h, t1);
    // 2) t3 = conv(t1, w3): 32ch, 3 height taps dil 2 -> 256ch
    conv_gemm_wmma<C8, 3, C, 0, 2, 2, float, float>
        <<<GRID_O256, 256, 0, stream>>>(t1, w3h, t3);
    // 3) t4 = conv(t3, w4): 256 -> 1, width taps dil 3
    t4_kernel<<<(NB * HW + 255) / 256, 256, 0, stream>>>(t3, w4, t4);
    // 4) t5 = depthwise(p2w * t1)  (f16 out, GEMM-only consumer)
    t5_kernel<<<(NB * C8 * HW + 255) / 256, 256, 0, stream>>>(t1, p2w, w5, t5h);
    // 5) t8 = conv(t5, w8): 32ch, 7 width taps dil 2 -> 32ch (f16 in/out)
    conv_gemm_wmma<C8, 7, C8, 2, 0, 8, _Float16, _Float16>
        <<<GRID_O32, 256, 0, stream>>>(t5h, w8h, t8h);
    // 6) t11 = pooled/relu/averaged t4
    pool_kernel<<<(NB * HW + 255) / 256, 256, 0, stream>>>(t4, t11);
    // 7) t14 = 3x3 conv of t11
    t14_kernel<<<(NB * HW + 255) / 256, 256, 0, stream>>>(t11, w14, t14);
    // 8) t12 = conv(t8, w12): 32ch, 7 width taps dil 1 -> 256ch (f16 in/out)
    conv_gemm_wmma<C8, 7, C, 1, 0, 2, _Float16, _Float16>
        <<<GRID_O256, 256, 0, stream>>>(t8h, w12h, t12h);
    // 9) t15 = conv(t12, w15'): fused t13-stack + 1x1 == 3 height taps dil 2
    conv_gemm_wmma<C, 3, C, 0, 2, 2, _Float16, float>
        <<<GRID_O256, 256, 0, stream>>>(t12h, w15h, t15);
    // 10) out = t14 + depthwise(t15, w16)
    final_kernel<<<(NB * C * HW + 255) / 256, 256, 0, stream>>>(t15, w16, t14, out);
}
